// CoAttention_85882166051617
// MI455X (gfx1250) — compile-verified, matrix operand order checked by reading, not run on hardware
//
#include <hip/hip_runtime.h>
#include <hip/hip_bf16.h>

// -------- CDNA5 (gfx1250, wave32) WMMA types --------
typedef __attribute__((ext_vector_type(16))) __bf16 v16bf;
typedef __attribute__((ext_vector_type(8)))  float  v8f;
typedef __attribute__((ext_vector_type(4)))  int    v4i;

typedef __attribute__((address_space(1))) v4i as1_v4i;
typedef __attribute__((address_space(3))) v4i as3_v4i;

// Async global->LDS staging (gfx1250). Guarded: falls back to vector copies.
#if __has_builtin(__builtin_amdgcn_global_load_async_to_lds_b128)
#define NL_ASYNC 1
#else
#define NL_ASYNC 0
#endif

__device__ __forceinline__ void nl_wait_async() {
#if NL_ASYNC
#if __has_builtin(__builtin_amdgcn_s_wait_asynccnt)
  __builtin_amdgcn_s_wait_asynccnt(0);
#else
  asm volatile("s_wait_asynccnt 0x0" ::: "memory");
#endif
#endif
}

__device__ __forceinline__ __bf16 f32_to_bf16(float f) {
  unsigned u = __builtin_bit_cast(unsigned, f);
  unsigned r = u + 0x7FFFu + ((u >> 16) & 1u);   // round-to-nearest-even
  unsigned short h = (unsigned short)(r >> 16);
  return __builtin_bit_cast(__bf16, h);
}

// Copy 32 B (16 bf16) global -> LDS: async-to-LDS when available.
__device__ __forceinline__ void nl_stage32B(const __bf16* __restrict__ g,
                                            __bf16* __restrict__ s) {
#if NL_ASYNC
  __builtin_amdgcn_global_load_async_to_lds_b128((as1_v4i*)g, (as3_v4i*)s, 0, 0);
  __builtin_amdgcn_global_load_async_to_lds_b128((as1_v4i*)g, (as3_v4i*)s, 16, 0);
#else
  ((uint4*)s)[0] = ((const uint4*)g)[0];
  ((uint4*)s)[1] = ((const uint4*)g)[1];
#endif
}

// -------- f32 -> bf16 conversion, 8 elems/thread, vectorized --------
__global__ void nl_cvt_bf16(const float* __restrict__ s, __bf16* __restrict__ d, long n8) {
  long i = (long)blockIdx.x * blockDim.x + threadIdx.x;
  long stride = (long)gridDim.x * blockDim.x;
  for (; i < n8; i += stride) {
    const float4 f0 = ((const float4*)s)[i * 2 + 0];
    const float4 f1 = ((const float4*)s)[i * 2 + 1];
    __bf16 o[8];
    o[0] = f32_to_bf16(f0.x); o[1] = f32_to_bf16(f0.y);
    o[2] = f32_to_bf16(f0.z); o[3] = f32_to_bf16(f0.w);
    o[4] = f32_to_bf16(f1.x); o[5] = f32_to_bf16(f1.y);
    o[6] = f32_to_bf16(f1.z); o[7] = f32_to_bf16(f1.w);
    ((uint4*)d)[i] = *(const uint4*)o;   // 16 B store of 8 bf16
  }
}

// -------- Generic batched bf16 GEMM via v_wmma_f32_16x16x32_bf16 --------
// Out[b, i, j] = alpha * sum_k A[b,i,k] * B[b,k,j] (+ bias[i]) (+ resid if FINAL)
// AK: A rows contiguous in k (as_k==1); else as_i==1 (transpose staging).
// BK: B contiguous in k (bs_k==1);     else bs_j==1 (transpose staging).
// Block tile 128x128, 256 threads = 8 wave32s; each wave: 2x4 subtiles of 16x16.
template <bool AK, bool BK, bool FINAL>
__global__ __launch_bounds__(256) void nl_gemm_wmma(
    const __bf16* __restrict__ A, long as_i, long as_k, long a_bat,
    const __bf16* __restrict__ B, long bs_k, long bs_j, long b_bat,
    const float* __restrict__ bias,
    const float* __restrict__ resid, long r_bat,
    void* __restrict__ OutP, long out_ld, long out_bat,
    int K, float alpha)
{
  constexpr int LDS_LD = 48;                     // 96 B row stride -> 32 B aligned halves
  __shared__ alignas(32) __bf16 sA[128 * LDS_LD];
  __shared__ alignas(32) __bf16 sB[128 * LDS_LD];

  const int  t     = threadIdx.x;
  const long batch = blockIdx.z;
  const long tileM = (long)blockIdx.x * 128;
  const long tileN = (long)blockIdx.y * 128;

  const __bf16* Ab = A + batch * a_bat;
  const __bf16* Bb = B + batch * b_bat;

  // wave arrangement: 4 (M) x 2 (N)
  const int lane  = t & 31;
  const int w     = t >> 5;
  const int row0  = (w >> 1) * 32;     // wave rows [row0, row0+32)
  const int col0  = (w & 1) * 64;      // wave cols [col0, col0+64)
  const int lrow  = lane & 15;
  const int khalf = (lane >> 4) << 4;  // lanes 0-15 -> K 0..15, lanes 16-31 -> K 16..31

  const v8f vzero = {0.f, 0.f, 0.f, 0.f, 0.f, 0.f, 0.f, 0.f};
  v8f acc[2][4];
#pragma unroll
  for (int sm = 0; sm < 2; ++sm)
#pragma unroll
    for (int sn = 0; sn < 4; ++sn) acc[sm][sn] = vzero;

  // staging roles
  const int srow = t & 127;            // row/col index, K-contiguous mode
  const int skc  = (t >> 7) << 4;      // 0 or 16
  const int tk0  = (t & 15) * 2;       // k pair, transpose mode
  const int tf0  = (t >> 4) * 8;       // fast-dim chunk, transpose mode

  for (int kk = 0; kk < K; kk += 32) {
    // ---- stage A tile: sA[i][k] = A[tileM+i, kk+k] ----
    if constexpr (AK) {                // rows contiguous in K: 32 B (async) path
      const __bf16* g = Ab + (tileM + srow) * as_i + (kk + skc);
      nl_stage32B(g, &sA[srow * LDS_LD + skc]);
      if (kk + 32 < K) __builtin_prefetch(g + 32, 0, 3);
    } else {                           // as_i == 1: contiguous along i, transpose on LDS write
      const __bf16* g0 = Ab + (tileM + tf0) + (long)(kk + tk0) * as_k;
      uint4 r0 = *(const uint4*)g0;
      uint4 r1 = *(const uint4*)(g0 + as_k);
      const unsigned short* a0 = (const unsigned short*)&r0;
      const unsigned short* a1 = (const unsigned short*)&r1;
#pragma unroll
      for (int e = 0; e < 8; ++e)
        *(unsigned*)&sA[(tf0 + e) * LDS_LD + tk0] =
            (unsigned)a0[e] | ((unsigned)a1[e] << 16);
      if (kk + 32 < K) __builtin_prefetch(g0 + 32 * as_k, 0, 3);
    }
    // ---- stage B tile (transposed in LDS): sB[j][k] = B[kk+k, tileN+j] ----
    if constexpr (BK) {                // contiguous along k: 32 B (async) path
      const __bf16* g = Bb + (tileN + srow) * bs_j + (kk + skc);
      nl_stage32B(g, &sB[srow * LDS_LD + skc]);
      if (kk + 32 < K) __builtin_prefetch(g + 32, 0, 3);
    } else {                           // bs_j == 1: contiguous along j, transpose on LDS write
      const __bf16* g0 = Bb + (long)(kk + tk0) * bs_k + (tileN + tf0);
      uint4 r0 = *(const uint4*)g0;
      uint4 r1 = *(const uint4*)(g0 + bs_k);
      const unsigned short* b0 = (const unsigned short*)&r0;
      const unsigned short* b1 = (const unsigned short*)&r1;
#pragma unroll
      for (int e = 0; e < 8; ++e)
        *(unsigned*)&sB[(tf0 + e) * LDS_LD + tk0] =
            (unsigned)b0[e] | ((unsigned)b1[e] << 16);
      if (kk + 32 < K) __builtin_prefetch(g0 + 32 * bs_k, 0, 3);
    }
    if constexpr (AK || BK) nl_wait_async();   // drain ASYNCcnt before publishing LDS
    __syncthreads();

    v16bf af[2], bfr[4];
#pragma unroll
    for (int sm = 0; sm < 2; ++sm)
      af[sm] = *(const v16bf*)&sA[(row0 + sm * 16 + lrow) * LDS_LD + khalf];
#pragma unroll
    for (int sn = 0; sn < 4; ++sn)
      bfr[sn] = *(const v16bf*)&sB[(col0 + sn * 16 + lrow) * LDS_LD + khalf];

#pragma unroll
    for (int sm = 0; sm < 2; ++sm)
#pragma unroll
      for (int sn = 0; sn < 4; ++sn)
        acc[sm][sn] = __builtin_amdgcn_wmma_f32_16x16x32_bf16(
            /*neg_a=*/false, af[sm], /*neg_b=*/false, bfr[sn],
            /*c_mod=*/(short)0, acc[sm][sn],
            /*reuse_a=*/false, /*reuse_b=*/false);

    __syncthreads();
  }

  // Epilogue. C/D layout: VGPR r <-> row (r + 8*(lane>>4)), col = lane&15.
  const int rhi = (lane >> 4) << 3;
#pragma unroll
  for (int sm = 0; sm < 2; ++sm) {
#pragma unroll
    for (int sn = 0; sn < 4; ++sn) {
#pragma unroll
      for (int r = 0; r < 8; ++r) {
        const long row = tileM + row0 + sm * 16 + r + rhi;
        const long col = tileN + col0 + sn * 16 + (lane & 15);
        float v = acc[sm][sn][r] * alpha;
        if (bias) v += bias[row];
        const long oidx = batch * out_bat + row * out_ld + col;
        if constexpr (FINAL) {
          const float* R = resid + batch * r_bat;
          ((float*)OutP)[oidx] = v + R[row * out_ld + col];
        } else {
          ((__bf16*)OutP)[oidx] = f32_to_bf16(v);
        }
      }
    }
  }
}

extern "C" void kernel_launch(void* const* d_in, const int* in_sizes, int n_in,
                              void* d_out, int out_size, void* d_ws, size_t ws_size,
                              hipStream_t stream) {
  (void)in_sizes; (void)n_in; (void)out_size; (void)ws_size;
  const long Bc = 8, C = 1024, N = 64L * 64, M = 16L * 16, Ci = 512;

  const float* x       = (const float*)d_in[0];  // [8,1024,64,64]
  const float* target  = (const float*)d_in[1];  // [8,1024,16,16]
  const float* w_g     = (const float*)d_in[2];  // [512,1024]
  const float* b_g     = (const float*)d_in[3];
  const float* w_theta = (const float*)d_in[4];
  const float* b_theta = (const float*)d_in[5];
  const float* w_phi   = (const float*)d_in[6];
  const float* b_phi   = (const float*)d_in[7];
  const float* w_out   = (const float*)d_in[8];  // [1024,512]
  const float* b_out   = (const float*)d_in[9];

  // Workspace layout (bf16), ~156 MB total
  __bf16* p = (__bf16*)d_ws;
  __bf16* xb   = p; p += Bc * C * N;
  __bf16* tb   = p; p += Bc * C * M;
  __bf16* wgb  = p; p += Ci * C;
  __bf16* wthb = p; p += Ci * C;
  __bf16* wphb = p; p += Ci * C;
  __bf16* wob  = p; p += C * Ci;
  __bf16* Gb   = p; p += Bc * Ci * M;
  __bf16* PHb  = p; p += Bc * Ci * M;
  __bf16* THb  = p; p += Bc * Ci * N;
  __bf16* Fb   = p; p += Bc * N * M;
  __bf16* Yb   = p; p += Bc * Ci * N;

  dim3 blk(256);
  auto cvt = [&](const float* s, __bf16* d, long n) {
    long n8 = n / 8;
    long nb = (n8 + 255) / 256; if (nb > 2048) nb = 2048;
    nl_cvt_bf16<<<dim3((unsigned)nb), blk, 0, stream>>>(s, d, n8);
  };
  cvt(x, xb, Bc * C * N);
  cvt(target, tb, Bc * C * M);
  cvt(w_g, wgb, Ci * C);
  cvt(w_theta, wthb, Ci * C);
  cvt(w_phi, wphb, Ci * C);
  cvt(w_out, wob, C * Ci);

  // 1) G = Wg x T + b_g         [Ci, M] per batch       (A k-contig, B j-contig)
  nl_gemm_wmma<true, false, false><<<dim3(Ci/128, M/128, Bc), blk, 0, stream>>>(
      wgb, C, 1, 0,              tb, M, 1, C * M,
      b_g, nullptr, 0,           Gb, M, Ci * M, (int)C, 1.0f);

  // 2) PH = Wphi x T + b_phi    [Ci, M]
  nl_gemm_wmma<true, false, false><<<dim3(Ci/128, M/128, Bc), blk, 0, stream>>>(
      wphb, C, 1, 0,             tb, M, 1, C * M,
      b_phi, nullptr, 0,         PHb, M, Ci * M, (int)C, 1.0f);

  // 3) TH = Wtheta x X + b_th   [Ci, N]
  nl_gemm_wmma<true, false, false><<<dim3(Ci/128, N/128, Bc), blk, 0, stream>>>(
      wthb, C, 1, 0,             xb, N, 1, C * N,
      b_theta, nullptr, 0,       THb, N, Ci * N, (int)C, 1.0f);

  // 4) F = TH^T x PH / 256      [N, M]; A i-contig (transpose staging), B j-contig
  nl_gemm_wmma<false, false, false><<<dim3(N/128, M/128, Bc), blk, 0, stream>>>(
      THb, 1, N, Ci * N,         PHb, M, 1, Ci * M,
      nullptr, nullptr, 0,       Fb, M, N * M, (int)Ci, 1.0f / (float)M);

  // 5) Y = G x F^T              [Ci, N]; A k-contig, B k-contig (F^T)
  nl_gemm_wmma<true, true, false><<<dim3(Ci/128, N/128, Bc), blk, 0, stream>>>(
      Gb, M, 1, Ci * M,          Fb, 1, M, N * M,
      nullptr, nullptr, 0,       Yb, N, Ci * N, (int)M, 1.0f);

  // 6) Out = Wout x Y + b_out + x   [C, N], f32 output with residual
  nl_gemm_wmma<true, false, true><<<dim3(C/128, N/128, Bc), blk, 0, stream>>>(
      wob, Ci, 1, 0,             Yb, N, 1, Ci * N,
      b_out, x, C * N,           d_out, N, C * N, (int)Ci, 1.0f);
}
